// NSE_15625091023330
// MI455X (gfx1250) — matrix operand order repaired and checked by reading, not compiled, not run on hardware
//
#include <hip/hip_runtime.h>
#include <hip/hip_bf16.h>
#include <math.h>

// ---------------------------------------------------------------------------
// NSE forward: reader LSTM -> attention/compose/writer recurrence.
// B=64, L=1024, D=512.  All GEMMs are K=512 dual-GEMMs on v_wmma_f32_16x16x32_f16.
// Weights pre-packed into per-lane WMMA-B layout; A-panels staged to LDS by the
// Tensor Data Mover; each wave computes two adjacent 16x16 tiles sharing its
// A operand (reuse_a hint).  mem tensor (134MB) streams through the 192MB L2.
// ---------------------------------------------------------------------------

typedef __attribute__((ext_vector_type(16))) _Float16 v16h;
typedef __attribute__((ext_vector_type(8)))  float    v8f;

#define DEV static __device__ __forceinline__
#define GK 512              // K of every GEMM operand in this model
#define KT (GK / 32)        // 16 k-steps per operand

DEV float hsig(float v) { return fminf(fmaxf(0.2f * v + 0.5f, 0.0f), 1.0f); }

// A-matrix 16x32 f16 tile from a row-major panel (lda elems).
// ISA 7.12.2: lane -> M = (lane&15); lanes 0-15 carry K {0..7,16..23},
// lanes 16-31 carry K {8..15,24..31}.  Per lane: two contiguous 16B spans.
DEV v16h load_a_tile(const _Float16* A, int lda, int k0, int lane) {
    int row = lane & 15;
    int kb  = (lane >> 4) * 8;
    const _Float16* p = A + (size_t)row * (size_t)lda + k0;
    v16h a;
#pragma unroll
    for (int v = 0; v < 8; ++v) {
        int k = (v < 4) ? (kb + 2 * v) : (16 + kb + 2 * (v - 4));
        a[2 * v]     = p[k];
        a[2 * v + 1] = p[k + 1];
    }
    return a;
}

// ---------------------------------------------------------------------------
// Tensor Data Mover staging of a 16 x GK f16 panel into LDS.
// D# per CDNA5 ISA (2D tile, data_size=2B).
// This toolchain: 6-arg builtin (v4u, v8i, v4i, v4i, v8i, i32 cpol).
// ---------------------------------------------------------------------------
#if __has_builtin(__builtin_amdgcn_tensor_load_to_lds) && __has_builtin(__builtin_amdgcn_s_wait_tensorcnt)
#define NSE_USE_TDM 1
typedef __attribute__((ext_vector_type(4))) unsigned int v4u;
typedef __attribute__((ext_vector_type(8))) int          v8i;
typedef __attribute__((ext_vector_type(4))) int          v4i;

DEV void tdm_load_panel(const _Float16* lds_dst, const _Float16* gsrc, int strideElems) {
    unsigned long long ga = (unsigned long long)(uintptr_t)gsrc;
    unsigned int ldsOff   = (unsigned int)(uintptr_t)lds_dst;  // low 32b = LDS offset
    v4u g0;
    g0[0] = 1u;                                            // count=1 (valid user D#)
    g0[1] = ldsOff;                                        // lds_addr (bytes)
    g0[2] = (unsigned int)(ga & 0xFFFFFFFFu);              // global_addr[31:0]
    g0[3] = (unsigned int)((ga >> 32) & 0x01FFFFFFu) | (2u << 30); // addr[56:32]|type=2
    v8i g1;
    unsigned int td0 = (unsigned int)GK;                   // tensor_dim0
    unsigned int td1 = 16u;                                // tensor_dim1
    g1[0] = (int)(1u << 16);                               // data_size = 2 bytes
    g1[1] = (int)((td0 & 0xFFFFu) << 16);                  // tensor_dim0[15:0]
    g1[2] = (int)((td0 >> 16) | ((td1 & 0xFFFFu) << 16));  // td0[31:16] | td1[15:0]
    g1[3] = (int)((td1 >> 16) | ((unsigned)GK << 16));     // td1[31:16] | tile_dim0
    g1[4] = (int)16;                                       // tile_dim1=16, tile_dim2=0
    g1[5] = (int)(unsigned)strideElems;                    // tensor_dim0_stride[31:0]
    g1[6] = 0;
    g1[7] = 0;
    v4i g2 = {0, 0, 0, 0};                                 // 2D tensor: groups 2/3 unused
    v4i g3 = {0, 0, 0, 0};
    v8i g4 = {0, 0, 0, 0, 0, 0, 0, 0};                     // trailing group (unused)
    __builtin_amdgcn_tensor_load_to_lds(g0, g1, g2, g3, g4, 0);
}
#endif

// ---------------------------------------------------------------------------
// Z[MxN] = A1[MxGK]*B1 + A2[MxGK]*B2 + bias ; N compile-time so packed-B
// strides fold into immediate offsets.  One wave per PAIR of adjacent 16x16
// tiles: the A operand is loaded once and feeds two v_wmma (reuse_a hint).
// 8 waves/block share one TDM-staged M-row panel pair.
// ---------------------------------------------------------------------------
template <int N>
__global__ void __launch_bounds__(256)
nse_dual_gemm_f16(const _Float16* __restrict__ A1, int lda1, const v16h* __restrict__ PB1,
                  const _Float16* __restrict__ A2, int lda2, const v16h* __restrict__ PB2,
                  const float* __restrict__ bias,
                  float* __restrict__ outF, _Float16* __restrict__ outH)
{
    constexpr int tilesN = N / 16;
    constexpr size_t BSTEP = (size_t)tilesN * 32;          // v16h per k-tile row
    __shared__ __align__(128) _Float16 ldsA[2][16 * GK];   // 32KB of 320KB WGP LDS
    int lane   = threadIdx.x & 31;
    int wave   = threadIdx.x >> 5;
    int pairId = blockIdx.x * 8 + wave;                    // 2 n-tiles per pair
    int tm     = (blockIdx.x * 16) / tilesN;               // block-uniform (16|tilesN)
    int tn     = (pairId * 2) % tilesN;
    int m0     = tm * 16;

#ifdef NSE_USE_TDM
    if (wave == 0) {
        tdm_load_panel(&ldsA[0][0], A1 + (size_t)m0 * (size_t)lda1, lda1);
        tdm_load_panel(&ldsA[1][0], A2 + (size_t)m0 * (size_t)lda2, lda2);
        __builtin_amdgcn_s_wait_tensorcnt(0);
    }
    __syncthreads();
#else
    for (int c = threadIdx.x; c < 16 * (GK / 8); c += 256) {
        int r = c >> 6, q = c & 63;
        ((uint4*)&ldsA[0][0])[c] = *(const uint4*)(A1 + (size_t)(m0 + r) * lda1 + q * 8);
        ((uint4*)&ldsA[1][0])[c] = *(const uint4*)(A2 + (size_t)(m0 + r) * lda2 + q * 8);
    }
    __syncthreads();
#endif

    const v16h* pb1 = PB1 + (size_t)tn * 32 + lane;
    const v16h* pb2 = PB2 + (size_t)tn * 32 + lane;

    v8f acc0 = {};
    v8f acc1 = {};
#pragma unroll 4
    for (int kt = 0; kt < KT; ++kt) {
        if (kt + 1 < KT) {                                 // WGP-scope prefetch (loc=3)
            __builtin_prefetch(pb1 + (size_t)(kt + 1) * BSTEP, 0, 3);
            __builtin_prefetch(pb2 + (size_t)(kt + 1) * BSTEP, 0, 3);
        }
        v16h a1  = load_a_tile(&ldsA[0][0], GK, kt * 32, lane);
        v16h b1a = pb1[(size_t)kt * BSTEP];
        v16h b1b = pb1[(size_t)kt * BSTEP + 32];
        // same A operand for both tiles: cache via reuse_a hint
        acc0 = __builtin_amdgcn_wmma_f32_16x16x32_f16(false, a1, false, b1a,
                                                      (short)0, acc0, true,  false);
        acc1 = __builtin_amdgcn_wmma_f32_16x16x32_f16(false, a1, false, b1b,
                                                      (short)0, acc1, false, false);
        v16h a2  = load_a_tile(&ldsA[1][0], GK, kt * 32, lane);
        v16h b2a = pb2[(size_t)kt * BSTEP];
        v16h b2b = pb2[(size_t)kt * BSTEP + 32];
        acc0 = __builtin_amdgcn_wmma_f32_16x16x32_f16(false, a2, false, b2a,
                                                      (short)0, acc0, true,  false);
        acc1 = __builtin_amdgcn_wmma_f32_16x16x32_f16(false, a2, false, b2b,
                                                      (short)0, acc1, false, false);
    }

    int col0 = tn * 16 + (lane & 15);
    float bv0 = bias ? bias[col0]      : 0.0f;
    float bv1 = bias ? bias[col0 + 16] : 0.0f;
    int rowbase = m0 + ((lane >> 4) << 3);
#pragma unroll
    for (int r = 0; r < 8; ++r) {
        float v0 = acc0[r] + bv0;
        float v1 = acc1[r] + bv1;
        size_t o = (size_t)(rowbase + r) * (size_t)N + col0;
        if (outF) { outF[o] = v0; outF[o + 16] = v1; }
        if (outH) { outH[o] = (_Float16)v0; outH[o + 16] = (_Float16)v1; }
    }
}

// ---------------------------------------------------------------------------
// Pack a K x N row-major f32 weight into per-lane WMMA-B tiles (f16).
// Packed element i: tile = i>>9, lane = (i>>4)&31, e = i&15;
//   col = tn*16 + (lane&15); k = kt*32 + (e&8)*2 + (lane>>4)*8 + (e&7)
// ---------------------------------------------------------------------------
__global__ void nse_pack_b(const float* __restrict__ src, _Float16* __restrict__ dst,
                           int K, int N)
{
    size_t i = (size_t)blockIdx.x * blockDim.x + threadIdx.x;
    if (i >= (size_t)K * N) return;
    size_t tile = i >> 9;
    int lane = (int)((i >> 4) & 31);
    int e    = (int)(i & 15);
    int tilesN = N >> 4;
    int kt = (int)(tile / tilesN);
    int tn = (int)(tile - (size_t)kt * tilesN);
    int col = tn * 16 + (lane & 15);
    int k   = kt * 32 + ((e & 8) * 2) + ((lane >> 4) * 8) + (e & 7);
    dst[i] = (_Float16)src[(size_t)k * N + col];
}

// gates z (B x 4D, order i,f,g,o) -> update c in place, emit h (f32 + f16)
__global__ void nse_lstm_pointwise(const float* __restrict__ z,
                                   float* __restrict__ c,
                                   float* __restrict__ hF,
                                   _Float16* __restrict__ hH,
                                   int B, int D)
{
    int idx = blockIdx.x * blockDim.x + threadIdx.x;
    if (idx >= B * D) return;
    int b = idx / D, d = idx - b * D;
    const float* zb = z + (size_t)b * 4 * D;
    float i = hsig(zb[d]);
    float f = hsig(zb[D + d]);
    float g = tanhf(zb[2 * D + d]);
    float o = hsig(zb[3 * D + d]);
    float cn = f * c[idx] + i * g;
    c[idx] = cn;
    float h = o * tanhf(cn);
    hF[idx] = h;
    hH[idx] = (_Float16)h;
}

// s[b,l] = dot(o_t[b,:], mem[b,l,:]) ; one wave per (b,l)
__global__ void __launch_bounds__(256)
nse_attn_scores(const float* __restrict__ o_t,
                const float* __restrict__ mem,
                float* __restrict__ s,
                int B, int L, int D)
{
    int lane = threadIdx.x & 31;
    int wid  = (int)((blockIdx.x * blockDim.x + threadIdx.x) >> 5);
    if (wid >= B * L) return;
    int b = wid / L, l = wid - b * L;
    const float4* mp = (const float4*)(mem + ((size_t)b * L + l) * D);
    const float4* op = (const float4*)(o_t + (size_t)b * D);
    float acc = 0.0f;
    for (int j = lane; j < (D >> 2); j += 32) {
        float4 m4 = mp[j], o4 = op[j];
        acc += m4.x * o4.x + m4.y * o4.y + m4.z * o4.z + m4.w * o4.w;
    }
#pragma unroll
    for (int off = 16; off; off >>= 1) acc += __shfl_xor(acc, off, 32);
    if (lane == 0) s[wid] = acc;
}

// in-place softmax over each length-L row; one block per row
__global__ void nse_softmax_rows(float* __restrict__ s, int L)
{
    __shared__ float red[256];
    int tid = threadIdx.x;
    float* row = s + (size_t)blockIdx.x * L;
    float mx = -3.4e38f;
    for (int l = tid; l < L; l += 256) mx = fmaxf(mx, row[l]);
    red[tid] = mx; __syncthreads();
    for (int st = 128; st; st >>= 1) {
        if (tid < st) red[tid] = fmaxf(red[tid], red[tid + st]);
        __syncthreads();
    }
    mx = red[0]; __syncthreads();
    float sum = 0.0f;
    for (int l = tid; l < L; l += 256) {
        float e = __expf(row[l] - mx);
        row[l] = e;
        sum += e;
    }
    red[tid] = sum; __syncthreads();
    for (int st = 128; st; st >>= 1) {
        if (tid < st) red[tid] += red[tid + st];
        __syncthreads();
    }
    float inv = 1.0f / red[0];
    for (int l = tid; l < L; l += 256) row[l] *= inv;
}

// m_rt[b,d] = sum_l z[b,l] * mem[b,l,d]
__global__ void __launch_bounds__(256)
nse_weighted_read(const float* __restrict__ zatt,
                  const float* __restrict__ mem,
                  _Float16* __restrict__ mrtH,
                  int L, int D)
{
    int b = blockIdx.y;
    int d = blockIdx.x * blockDim.x + threadIdx.x;
    if (d >= D) return;
    const float* zb = zatt + (size_t)b * L;
    const float* mb = mem + (size_t)b * L * D + d;
    float acc = 0.0f;
    for (int l = 0; l < L; ++l) acc += zb[l] * mb[(size_t)l * D];
    mrtH[(size_t)b * D + d] = (_Float16)acc;
}

// mem = mem*(1-z) + h*z
__global__ void nse_mem_update(float4* __restrict__ mem4,
                               const float* __restrict__ zatt,
                               const float4* __restrict__ hw4,
                               int L, int Dq, size_t total)
{
    size_t idx = (size_t)blockIdx.x * blockDim.x + threadIdx.x;
    if (idx >= total) return;
    size_t per_b = (size_t)L * Dq;
    int b = (int)(idx / per_b);
    size_t rem = idx - (size_t)b * per_b;
    int l  = (int)(rem / Dq);
    int dq = (int)(rem - (size_t)l * Dq);
    float zv = zatt[(size_t)b * L + l];
    float om = 1.0f - zv;
    float4 m = mem4[idx];
    float4 h = hw4[(size_t)b * Dq + dq];
    m.x = m.x * om + h.x * zv;
    m.y = m.y * om + h.y * zv;
    m.z = m.z * om + h.z * zv;
    m.w = m.w * om + h.w * zv;
    mem4[idx] = m;
}

__global__ void nse_f32_to_f16(const float* __restrict__ s, _Float16* __restrict__ d, size_t n)
{
    size_t i = (size_t)blockIdx.x * blockDim.x + threadIdx.x;
    if (i < n) d[i] = (_Float16)s[i];
}

__global__ void nse_copy_f32v4(const float4* __restrict__ s, float4* __restrict__ d, size_t n4)
{
    size_t i = (size_t)blockIdx.x * blockDim.x + threadIdx.x;
    if (i < n4) d[i] = s[i];
}

__global__ void nse_copy_f32(const float* __restrict__ s, float* __restrict__ d, size_t n)
{
    size_t i = (size_t)blockIdx.x * blockDim.x + threadIdx.x;
    if (i < n) d[i] = s[i];
}

__global__ void nse_zero_f32(float* __restrict__ d, size_t n)
{
    size_t i = (size_t)blockIdx.x * blockDim.x + threadIdx.x;
    if (i < n) d[i] = 0.0f;
}

__global__ void nse_zero_f16(_Float16* __restrict__ d, size_t n)
{
    size_t i = (size_t)blockIdx.x * blockDim.x + threadIdx.x;
    if (i < n) d[i] = (_Float16)0.0f;
}

extern "C" void kernel_launch(void* const* d_in, const int* in_sizes, int n_in,
                              void* d_out, int out_size, void* d_ws, size_t ws_size,
                              hipStream_t stream)
{
    (void)in_sizes; (void)n_in; (void)out_size; (void)ws_size;
    const int B = 64, L = 1024, D = 512;
    const int D4 = 4 * D;

    const float* x  = (const float*)d_in[0];
    const float* br = (const float*)d_in[3];
    const float* bw = (const float*)d_in[6];
    const float* bc = (const float*)d_in[8];

    // ---- workspace carve ----
    char* p = (char*)d_ws;
    auto alloc = [&](size_t bytes) -> char* {
        char* r = p;
        p += (bytes + 255) & ~(size_t)255;
        return r;
    };
    _Float16* xh     = (_Float16*)alloc((size_t)B * L * D * 2);   // x in f16
    v16h*     WrP    = (v16h*)    alloc((size_t)D * D4 * 2);      // packed weights
    v16h*     UrP    = (v16h*)    alloc((size_t)D * D4 * 2);
    v16h*     WwP    = (v16h*)    alloc((size_t)D * D4 * 2);
    v16h*     UwP    = (v16h*)    alloc((size_t)D * D4 * 2);
    v16h*     WcTopP = (v16h*)    alloc((size_t)D * D * 2);
    v16h*     WcBotP = (v16h*)    alloc((size_t)D * D * 2);
    float*    o_seq  = (float*)   alloc((size_t)L * B * D * 4);   // reader h (f32)
    _Float16* o_seqh = (_Float16*)alloc((size_t)L * B * D * 2);   // reader h (f16)
    float*    mem    = (float*)   alloc((size_t)B * L * D * 4);   // 134MB, L2-resident
    float*    zbuf   = (float*)   alloc((size_t)B * D4 * 4);
    float*    satt   = (float*)   alloc((size_t)B * L * 4);
    _Float16* mrt_h  = (_Float16*)alloc((size_t)B * D * 2);
    _Float16* ct_h   = (_Float16*)alloc((size_t)B * D * 2);
    float*    hw     = (float*)   alloc((size_t)B * D * 4);
    _Float16* hw_h   = (_Float16*)alloc((size_t)B * D * 2);
    float*    cr     = (float*)   alloc((size_t)B * D * 4);
    float*    cw     = (float*)   alloc((size_t)B * D * 4);
    _Float16* zeroh  = (_Float16*)alloc((size_t)B * D * 2);

    auto blk = [](size_t n, int t) { return (int)((n + t - 1) / t); };

    // ---- setup (re-run every call: deterministic) ----
    {
        size_t nx = (size_t)B * L * D;
        nse_f32_to_f16<<<blk(nx, 256), 256, 0, stream>>>(x, xh, nx);
        size_t nw = (size_t)D * D4;
        nse_pack_b<<<blk(nw, 256), 256, 0, stream>>>((const float*)d_in[1], (_Float16*)WrP, D, D4);
        nse_pack_b<<<blk(nw, 256), 256, 0, stream>>>((const float*)d_in[2], (_Float16*)UrP, D, D4);
        nse_pack_b<<<blk(nw, 256), 256, 0, stream>>>((const float*)d_in[4], (_Float16*)WwP, D, D4);
        nse_pack_b<<<blk(nw, 256), 256, 0, stream>>>((const float*)d_in[5], (_Float16*)UwP, D, D4);
        size_t nh = (size_t)D * D;
        const float* Wc = (const float*)d_in[7];
        nse_pack_b<<<blk(nh, 256), 256, 0, stream>>>(Wc,                 (_Float16*)WcTopP, D, D);
        nse_pack_b<<<blk(nh, 256), 256, 0, stream>>>(Wc + (size_t)D * D, (_Float16*)WcBotP, D, D);
        nse_copy_f32v4<<<blk(nx / 4, 256), 256, 0, stream>>>((const float4*)x, (float4*)mem, nx / 4);
        size_t nbd = (size_t)B * D;
        nse_zero_f32<<<blk(nbd, 256), 256, 0, stream>>>(cr, nbd);
        nse_zero_f32<<<blk(nbd, 256), 256, 0, stream>>>(cw, nbd);
        nse_zero_f32<<<blk(nbd, 256), 256, 0, stream>>>(hw, nbd);
        nse_zero_f16<<<blk(nbd, 256), 256, 0, stream>>>(hw_h, nbd);
        nse_zero_f16<<<blk(nbd, 256), 256, 0, stream>>>(zeroh, nbd);
    }

    // tile pairs per launch: big (M=64,N=2048): 512 tiles -> 256 pairs -> 32 blocks
    //                        comp (M=64,N=512): 128 tiles ->  64 pairs ->  8 blocks
    const int gemmBlocksBig  = 32;
    const int gemmBlocksComp = 8;

    // ---- reader LSTM scan ----
    for (int t = 0; t < L; ++t) {
        const _Float16* A1 = xh + (size_t)t * D;           // x_t rows, stride L*D
        const _Float16* A2 = (t == 0) ? zeroh : (o_seqh + (size_t)(t - 1) * B * D);
        nse_dual_gemm_f16<2048><<<gemmBlocksBig, 256, 0, stream>>>(
            A1, L * D, WrP,
            A2, D,     UrP,
            br, zbuf, (_Float16*)nullptr);
        nse_lstm_pointwise<<<blk((size_t)B * D, 256), 256, 0, stream>>>(
            zbuf, cr, o_seq + (size_t)t * B * D, o_seqh + (size_t)t * B * D, B, D);
    }

    // ---- writer recurrence over memory ----
    for (int t = 0; t < L; ++t) {
        const float*    o_t  = o_seq  + (size_t)t * B * D;
        const _Float16* o_th = o_seqh + (size_t)t * B * D;

        nse_attn_scores<<<(B * L) / 8, 256, 0, stream>>>(o_t, mem, satt, B, L, D);
        nse_softmax_rows<<<B, 256, 0, stream>>>(satt, L);
        nse_weighted_read<<<dim3(D / 256, B), 256, 0, stream>>>(satt, mem, mrt_h, L, D);

        // c_t = [o_t, m_rt] @ Wc + bc  (split-K dual GEMM over Wc halves)
        nse_dual_gemm_f16<512><<<gemmBlocksComp, 256, 0, stream>>>(
            o_th,  D, WcTopP,
            mrt_h, D, WcBotP,
            bc, (float*)nullptr, ct_h);

        // writer LSTM gates
        nse_dual_gemm_f16<2048><<<gemmBlocksBig, 256, 0, stream>>>(
            ct_h, D, WwP,
            hw_h, D, UwP,
            bw, zbuf, (_Float16*)nullptr);
        nse_lstm_pointwise<<<blk((size_t)B * D, 256), 256, 0, stream>>>(
            zbuf, cw, hw, hw_h, B, D);

        size_t total4 = (size_t)B * L * (D / 4);
        nse_mem_update<<<blk(total4, 256), 256, 0, stream>>>(
            (float4*)mem, satt, (const float4*)hw, L, D / 4, total4);
    }

    // return_mode == 'last_output': h_T of writer LSTM
    nse_copy_f32<<<blk((size_t)B * D, 256), 256, 0, stream>>>(hw, (float*)d_out, (size_t)B * D);
}
